// AdaptiveDWCBlock_66632122630636
// MI455X (gfx1250) — matrix operand order repaired and checked by reading, not compiled, not run on hardware
//
#include <hip/hip_runtime.h>
#include <math.h>

// Problem constants (match reference)
#define BB    2
#define CC    64
#define HH_   128
#define WW_   128
#define NN    (HH_ * WW_)      // 16384
#define KK    16
#define HD    32
#define RANK  8
#define MLPD  32
#define DD    (CC * 9)         // 576
#define KCH   (DD / 32)        // 18 k-chunks of 32 for f16 WMMA
#define SPW_  10.0f
#define EPS_  1e-6f

typedef __attribute__((ext_vector_type(16))) _Float16 v16h;
typedef __attribute__((ext_vector_type(8)))  float    v8f;

// ---------------------------------------------------------------------------
// 3x3 patch gather with zero padding. d = c*9 + (3*i + j), offsets i,j in 0..2
// ---------------------------------------------------------------------------
__device__ __forceinline__ float patch_val(const float* __restrict__ x, int b,
                                           int cin, int h, int w, int d) {
  int c  = d / 9;
  int t  = d - c * 9;
  int ph = h + t / 3 - 1;
  int pw = w + (t - (t / 3) * 3) - 1;
  if ((unsigned)ph >= (unsigned)HH_ || (unsigned)pw >= (unsigned)WW_) return 0.f;
  return x[(((size_t)b * cin + c) * HH_ + ph) * WW_ + pw];
}

// A-fragment packing (16-bit A 16x32, ISA 7.12.2):
//   lanes 0-15 : row M=lane, halves 0..7 -> K 0..7,  halves 8..15 -> K 16..23
//   lanes 16-31: row M=lane-16, halves  -> K 8..15 and K 24..31
__device__ __forceinline__ void a_pack_pos(int mm, int kk, int& lane, int& hh) {
  lane = mm + (((kk & 15) >= 8) ? 16 : 0);
  hh   = (kk & 7) + ((kk >= 16) ? 8 : 0);
}

// B-fragment packing (K x 16, same K order as A):
//   lanes 0-15 hold column N=lane with K 0..7 in halves 0..7, K 16..23 in halves 8..15;
//   lanes 16-31 hold same columns with K 8..15 and K 24..31.
__device__ __forceinline__ void b_pack_pos(int kk, int col, int& lane, int& hh) {
  lane = (col & 15) + (((kk & 15) >= 8) ? 16 : 0);
  hh   = (kk & 7) + ((kk >= 16) ? 8 : 0);
}

// ---------------------------------------------------------------------------
// Pack conv weights (OIHW, BN folded) into WMMA B-fragment order (f16)
// ---------------------------------------------------------------------------
__global__ void convpack_kernel(const float* __restrict__ w, const float* __restrict__ cb,
                                const float* __restrict__ g, const float* __restrict__ bt,
                                const float* __restrict__ mm, const float* __restrict__ vv,
                                int CIN, int COUT, _Float16* __restrict__ wp,
                                float* __restrict__ bias2) {
  int KD  = CIN * 9;
  int idx = blockIdx.x * 256 + threadIdx.x;
  if (idx < COUT) {
    float so   = g[idx] * rsqrtf(vv[idx] + 1e-5f);
    bias2[idx] = (cb[idx] - mm[idx]) * so + bt[idx];
  }
  if (idx >= KD * COUT) return;
  int d = idx / COUT, o = idx - (idx / COUT) * COUT;
  int c = d / 9, t = d - (d / 9) * 9;
  float so  = g[o] * rsqrtf(vv[o] + 1e-5f);
  float val = w[(o * CIN + c) * 9 + t] * so;
  int kc = d >> 5, kk = d & 31;
  int nt = o >> 4;
  int lane, hh;
  b_pack_pos(kk, o, lane, hh);
  int NT = COUT / 16;
  wp[((kc * NT + nt) * 32 + lane) * 16 + hh] = (_Float16)val;
}

// ---------------------------------------------------------------------------
// 3x3 conv (pad 1) as implicit GEMM on WMMA, BN folded into weights, PReLU.
// One block = 32 pixels (2 M-subtiles); 4 waves = (msub, ntile) pairs.
// ---------------------------------------------------------------------------
template <int CIN, int COUT>
__global__ __launch_bounds__(128)
void conv3x3_wmma(const float* __restrict__ in, const _Float16* __restrict__ wp,
                  const float* __restrict__ bias2, const float* __restrict__ aptr,
                  float* __restrict__ out) {
  constexpr int KD  = CIN * 9;
  constexpr int NCH = KD / 32;
  constexpr int NT  = COUT / 16;                // 2
  __shared__ __align__(32) _Float16 Alds[2 * NCH * 32 * 16];
  const int b   = blockIdx.y;
  const int m0  = blockIdx.x * 32;
  const int tid = threadIdx.x;

  for (int e = tid; e < 32 * (KD / 2); e += 128) {
    int mm = e / (KD / 2);
    int dp = (e - mm * (KD / 2)) * 2;
    int n  = m0 + mm;
    int h  = n / WW_, w = n - (n / WW_) * WW_;
    float v0 = patch_val(in, b, CIN, h, w, dp);
    float v1 = patch_val(in, b, CIN, h, w, dp + 1);
    int kc = dp >> 5, kk = dp & 31;
    int msub = mm >> 4, mrow = mm & 15;
    int lane, hh;
    a_pack_pos(mrow, kk, lane, hh);             // kk even -> hh even, pair contiguous
    _Float16* p = &Alds[((msub * NCH + kc) * 32 + lane) * 16 + hh];
    p[0] = (_Float16)v0;
    p[1] = (_Float16)v1;
  }
  __syncthreads();

  const int wv = tid >> 5, lane = tid & 31;
  const int msub = wv >> 1, nt = wv & 1;
  v8f acc = {};
  for (int kc = 0; kc < NCH; ++kc) {
    v16h a  = *(const v16h*)&Alds[((msub * NCH + kc) * 32 + lane) * 16];
    v16h bm = *(const v16h*)&wp[((kc * NT + nt) * 32 + lane) * 16];
    acc = __builtin_amdgcn_wmma_f32_16x16x32_f16(false, a, false, bm, (short)0, acc,
                                                 false, false);
  }
  const int   col = nt * 16 + (lane & 15);
  const float bz  = bias2[col];
  const float al  = *aptr;
  const int   mhi = (lane >= 16) ? 8 : 0;
#pragma unroll
  for (int r = 0; r < 8; ++r) {
    int   n = m0 + msub * 16 + r + mhi;
    float y = acc[r] + bz;
    y       = (y > 0.f) ? y : al * y;           // PReLU
    out[(((size_t)b * COUT + col) * NN) + n] = y;
  }
}

// ---------------------------------------------------------------------------
// 1x1 conv (no BN/act): feats[b,n,o] = sum_c f2[b,c,n]*w[o,c] + b[o]
// ---------------------------------------------------------------------------
__global__ void conv1x1_kernel(const float* __restrict__ f2, const float* __restrict__ w3,
                               const float* __restrict__ b3, float* __restrict__ feats) {
  int idx = blockIdx.x * 256 + threadIdx.x;
  if (idx >= BB * NN * HD) return;
  int o = idx & 31;
  int n = (idx >> 5) & (NN - 1);
  int b = idx >> 19;                            // 32*NN = 2^19
  float s = b3[o];
#pragma unroll
  for (int c = 0; c < HD; ++c) s += f2[((size_t)(b * HD + c)) * NN + n] * w3[o * HD + c];
  feats[idx] = s;
}

// ---------------------------------------------------------------------------
// Soft k-means
// ---------------------------------------------------------------------------
__global__ void kmeans_init(const float* __restrict__ fcp, float* __restrict__ fc,
                            float* __restrict__ sc) {
  int i = blockIdx.x * 64 + threadIdx.x;
  if (i < BB * KK * HD) fc[i] = fcp[i % (KK * HD)];
  if (i < KK) {
    int gi = i / 4, gj = i % 4;                 // gh = gw = 4 for K=16
    sc[i * 2 + 0] = (gj + 0.5f) / 4.f * 2.f - 1.f;
    sc[i * 2 + 1] = (gi + 0.5f) / 4.f * 2.f - 1.f;
  }
}

__global__ __launch_bounds__(256)
void kmeans_assign(const float* __restrict__ feats, const float* __restrict__ fc,
                   const float* __restrict__ sc, float* __restrict__ sa,
                   int* __restrict__ labels, int write_labels) {
  int idx = blockIdx.x * 256 + threadIdx.x;
  if (idx >= BB * NN) return;
  int b = idx / NN, n = idx - b * NN;
  const float* f = feats + (size_t)idx * HD;
  float fv[HD], fn2 = 0.f;
#pragma unroll
  for (int d = 0; d < HD; ++d) { fv[d] = f[d]; fn2 += fv[d] * fv[d]; }
  int   h  = n / WW_, w = n - (n / WW_) * WW_;
  float sx = -1.f + 2.f * w / (WW_ - 1);
  float sy = -1.f + 2.f * h / (HH_ - 1);
  float lg[KK], mx = -3.4e38f;
  for (int k = 0; k < KK; ++k) {
    const float* cp = fc + ((size_t)b * KK + k) * HD;
    float cn2 = 0.f, dt = 0.f;
#pragma unroll
    for (int d = 0; d < HD; ++d) { float c = cp[d]; cn2 += c * c; dt += c * fv[d]; }
    float fd = sqrtf(fmaxf(fn2 + cn2 - 2.f * dt, 0.f));
    float dx = sx - sc[k * 2], dy = sy - sc[k * 2 + 1];
    float sd = sqrtf(dx * dx + dy * dy);
    lg[k] = -(fd + SPW_ * sd);                  // TEMP = 1
    mx    = fmaxf(mx, lg[k]);
  }
  float s = 0.f;
#pragma unroll
  for (int k = 0; k < KK; ++k) { lg[k] = expf(lg[k] - mx); s += lg[k]; }
  float  inv = 1.f / s;
  int    best = 0;
  float  bv = -1.f;
  float* so = sa + (size_t)idx * KK;
#pragma unroll
  for (int k = 0; k < KK; ++k) {
    float p = lg[k] * inv;
    so[k]   = p;
    if (p > bv) { bv = p; best = k; }
  }
  if (write_labels) labels[idx] = best;
}

__global__ __launch_bounds__(256)
void reduce_fc_kernel(const float* __restrict__ sa, const float* __restrict__ feats,
                      float* __restrict__ fc) {
  int bk = blockIdx.x, b = bk / KK, k = bk - (bk / KK) * KK;
  float acc[HD], accw = 0.f;
#pragma unroll
  for (int d = 0; d < HD; ++d) acc[d] = 0.f;
  for (int n = threadIdx.x; n < NN; n += 256) {
    float s = sa[((size_t)(b * NN + n)) * KK + k];
    const float* f = feats + ((size_t)(b * NN + n)) * HD;
    accw += s;
#pragma unroll
    for (int d = 0; d < HD; ++d) acc[d] += s * f[d];
  }
  for (int o = 16; o > 0; o >>= 1) {
    accw += __shfl_down(accw, o, 32);
#pragma unroll
    for (int d = 0; d < HD; ++d) acc[d] += __shfl_down(acc[d], o, 32);
  }
  __shared__ float part[8][HD + 1];
  int wv = threadIdx.x >> 5, ln = threadIdx.x & 31;
  if (ln == 0) {
#pragma unroll
    for (int d = 0; d < HD; ++d) part[wv][d] = acc[d];
    part[wv][HD] = accw;
  }
  __syncthreads();
  if (threadIdx.x < HD + 1) {
    float s = 0.f;
    for (int w = 0; w < 8; ++w) s += part[w][threadIdx.x];
    part[0][threadIdx.x] = s;
  }
  __syncthreads();
  if (threadIdx.x < HD)
    fc[(size_t)bk * HD + threadIdx.x] = part[0][threadIdx.x] / (part[0][HD] + EPS_);
}

__global__ __launch_bounds__(256)
void reduce_sc_kernel(const float* __restrict__ sa, float* __restrict__ sc) {
  int   k = blockIdx.x;
  float s = 0.f, sx = 0.f, sy = 0.f;
  for (int idx = threadIdx.x; idx < BB * NN; idx += 256) {
    int   n = idx & (NN - 1);
    float v = sa[(size_t)idx * KK + k];
    int   h = n >> 7, w = n & 127;
    s  += v;
    sx += v * (-1.f + 2.f * w / (WW_ - 1));
    sy += v * (-1.f + 2.f * h / (HH_ - 1));
  }
  for (int o = 16; o > 0; o >>= 1) {
    s  += __shfl_down(s, o, 32);
    sx += __shfl_down(sx, o, 32);
    sy += __shfl_down(sy, o, 32);
  }
  __shared__ float part[8][3];
  int wv = threadIdx.x >> 5, ln = threadIdx.x & 31;
  if (ln == 0) { part[wv][0] = s; part[wv][1] = sx; part[wv][2] = sy; }
  __syncthreads();
  if (threadIdx.x == 0) {
    float S = 0.f, SX = 0.f, SY = 0.f;
    for (int i = 0; i < 8; ++i) { S += part[i][0]; SX += part[i][1]; SY += part[i][2]; }
    S /= BB; SX /= BB; SY /= BB;                // ws = sa.mean(0)
    float inv = 1.f / (S + EPS_);
    sc[k * 2 + 0] = SX * inv;
    sc[k * 2 + 1] = SY * inv;
  }
}

// ---------------------------------------------------------------------------
// Counting sort of pixels by label
// ---------------------------------------------------------------------------
__global__ void hist_kernel(const int* __restrict__ labels, int* __restrict__ counts) {
  int idx = blockIdx.x * 256 + threadIdx.x;
  if (idx >= BB * NN) return;
  atomicAdd(&counts[(idx / NN) * KK + labels[idx]], 1);
}

__global__ void offsets_kernel(const int* __restrict__ counts, int* __restrict__ offs,
                               int* __restrict__ cursor) {
  int b = threadIdx.x;
  if (b >= BB) return;
  int run = 0;
  for (int k = 0; k < KK; ++k) {
    offs[b * KK + k]   = run;
    cursor[b * KK + k] = run;
    run += counts[b * KK + k];
  }
}

__global__ void scatter_kernel(const int* __restrict__ labels, int* __restrict__ cursor,
                               int* __restrict__ order) {
  int idx = blockIdx.x * 256 + threadIdx.x;
  if (idx >= BB * NN) return;
  int b = idx / NN, n = idx - b * NN;
  int slot = atomicAdd(&cursor[b * KK + labels[idx]], 1);
  order[(size_t)b * NN + slot] = n;
}

// ---------------------------------------------------------------------------
// Cluster patch centers: centers[b,k,d] = sum_{n in k} patch(n,d) / (cnt+eps)
// ---------------------------------------------------------------------------
__global__ __launch_bounds__(192)
void centers_kernel(const float* __restrict__ x, const int* __restrict__ order,
                    const int* __restrict__ offs, const int* __restrict__ counts,
                    float* __restrict__ centers) {
  int bk = blockIdx.x, b = bk / KK;
  int cnt = counts[bk], ofs = offs[bk];
  int tid = threadIdx.x;
  __shared__ int pch[192];
  float a0 = 0.f, a1 = 0.f, a2 = 0.f;
  int d0 = tid, d1 = tid + 192, d2 = tid + 384;
  for (int p0 = 0; p0 < cnt; p0 += 192) {
    int e = (cnt - p0 < 192) ? (cnt - p0) : 192;
    if (tid < e) pch[tid] = order[(size_t)b * NN + ofs + p0 + tid];
    __syncthreads();
    for (int j = 0; j < e; ++j) {
      int n = pch[j];
      int h = n / WW_, w = n - (n / WW_) * WW_;
      a0 += patch_val(x, b, CC, h, w, d0);
      a1 += patch_val(x, b, CC, h, w, d1);
      a2 += patch_val(x, b, CC, h, w, d2);
    }
    __syncthreads();
  }
  float  inv = 1.f / ((float)cnt + EPS_);
  float* cp  = centers + (size_t)bk * DD;
  cp[d0] = a0 * inv;
  cp[d1] = a1 * inv;
  cp[d2] = a2 * inv;
}

// ---------------------------------------------------------------------------
// Per-(b,k) MLPs: low-rank coeffs lr[8] and bias[64]
// ---------------------------------------------------------------------------
__global__ __launch_bounds__(64)
void mlp_kernel(const float* __restrict__ centers, const float* __restrict__ lr1w,
                const float* __restrict__ lr1b, const float* __restrict__ lr2w,
                const float* __restrict__ lr2b, const float* __restrict__ lr3w,
                const float* __restrict__ lr3b, const float* __restrict__ bw1,
                const float* __restrict__ bb1, const float* __restrict__ bw2,
                const float* __restrict__ bb2, float* __restrict__ lrout,
                float* __restrict__ biasout) {
  int bk = blockIdx.x, tid = threadIdx.x;
  __shared__ float ctr[DD];
  __shared__ float h1[MLPD], hb[MLPD], h2[MLPD];
  for (int d = tid; d < DD; d += 64) ctr[d] = centers[(size_t)bk * DD + d];
  __syncthreads();
  if (tid < 32) {
    float s = lr1b[tid];
    for (int d = 0; d < DD; ++d) s += ctr[d] * lr1w[tid * DD + d];
    h1[tid] = fmaxf(s, 0.f);
  } else {
    int   j = tid - 32;
    float s = bb1[j];
    for (int d = 0; d < DD; ++d) s += ctr[d] * bw1[j * DD + d];
    hb[j] = fmaxf(s, 0.f);
  }
  __syncthreads();
  if (tid < 32) {
    float s = lr2b[tid];
#pragma unroll
    for (int j = 0; j < MLPD; ++j) s += h1[j] * lr2w[tid * MLPD + j];
    h2[tid] = fmaxf(s, 0.f);
  }
  __syncthreads();
  if (tid < RANK) {
    float s = lr3b[tid];
#pragma unroll
    for (int j = 0; j < MLPD; ++j) s += h2[j] * lr3w[tid * MLPD + j];
    lrout[bk * RANK + tid] = s;
  }
  {
    float s = bb2[tid];
#pragma unroll
    for (int j = 0; j < MLPD; ++j) s += hb[j] * bw2[tid * MLPD + j];
    biasout[bk * CC + tid] = s;
  }
}

// ---------------------------------------------------------------------------
// kern[b,k,d,c] = sum_r lr[b,k,r]*base[r,d,c], packed f16 in B-fragment order:
// layout [bk][kc][nt][lane][hh]
// ---------------------------------------------------------------------------
__global__ __launch_bounds__(256)
void kernpack_kernel(const float* __restrict__ lr, const float* __restrict__ base,
                     _Float16* __restrict__ kp) {
  int bk = blockIdx.x, kc = blockIdx.y, tid = threadIdx.x;
  __shared__ float l[RANK];
  if (tid < RANK) l[tid] = lr[bk * RANK + tid];
  __syncthreads();
  for (int e = tid; e < 32 * CC; e += 256) {
    int   kk = e >> 6, c = e & 63;
    int   d  = kc * 32 + kk;
    float v  = 0.f;
#pragma unroll
    for (int r = 0; r < RANK; ++r) v += l[r] * base[((size_t)r * DD + d) * CC + c];
    int nt = c >> 4;
    int lane, hh;
    b_pack_pos(kk, c, lane, hh);
    kp[((((size_t)bk * KCH + kc) * 4 + nt) * 32 + lane) * 16 + hh] = (_Float16)v;
  }
}

// ---------------------------------------------------------------------------
// Grouped gathered GEMM: per (b,k), tiles of 32 member pixels x 64 channels.
// A (32x576 f16) built in LDS in fragment order. 4 waves each own 16 output
// channels; per K-chunk one B fragment feeds TWO WMMAs (2 M-subtiles),
// halving B L2 traffic vs a 16-pixel tile.
// ---------------------------------------------------------------------------
__global__ __launch_bounds__(128)
void pwac_gemm(const float* __restrict__ x, const int* __restrict__ order,
               const int* __restrict__ offs, const int* __restrict__ counts,
               const _Float16* __restrict__ kp, const float* __restrict__ biasb,
               float* __restrict__ out) {
  const int b = blockIdx.z, k = blockIdx.y;
  const int bk  = b * KK + k;
  const int cnt = counts[bk];
  const int m0  = blockIdx.x * 32;
  if (m0 >= cnt) return;                        // early-exit empty tiles
  const int ofs = offs[bk];
  __shared__ int pix[32];
  __shared__ __align__(32) _Float16 Alds[2 * KCH * 32 * 16];  // 36 KB
  const int tid = threadIdx.x;
  if (tid < 32)
    pix[tid] = (m0 + tid < cnt) ? order[(size_t)b * NN + ofs + m0 + tid] : -1;
  __syncthreads();

  for (int e = tid; e < 32 * (DD / 2); e += 128) {
    int   mm = e / (DD / 2);
    int   dp = (e - mm * (DD / 2)) * 2;
    int   n  = pix[mm];
    float v0 = 0.f, v1 = 0.f;
    if (n >= 0) {
      int h = n / WW_, w = n - (n / WW_) * WW_;
      v0 = patch_val(x, b, CC, h, w, dp);
      v1 = patch_val(x, b, CC, h, w, dp + 1);
    }
    int kc = dp >> 5, kk = dp & 31;
    int msub = mm >> 4, mrow = mm & 15;
    int lane, hh;
    a_pack_pos(mrow, kk, lane, hh);
    _Float16* p = &Alds[((msub * KCH + kc) * 32 + lane) * 16 + hh];
    p[0] = (_Float16)v0;
    p[1] = (_Float16)v1;
  }
  __syncthreads();

  const int       wv  = tid >> 5, lane = tid & 31;   // wv = n-tile (0..3)
  const _Float16* kpb = kp + (size_t)bk * KCH * 4 * 512;
  v8f acc0 = {}, acc1 = {};
  for (int kc = 0; kc < KCH; ++kc) {
    v16h bm = *(const v16h*)&kpb[((kc * 4 + wv) * 32 + lane) * 16];
    v16h a0 = *(const v16h*)&Alds[(kc * 32 + lane) * 16];
    v16h a1 = *(const v16h*)&Alds[((KCH + kc) * 32 + lane) * 16];
    acc0 = __builtin_amdgcn_wmma_f32_16x16x32_f16(false, a0, false, bm, (short)0, acc0,
                                                  false, false);
    acc1 = __builtin_amdgcn_wmma_f32_16x16x32_f16(false, a1, false, bm, (short)0, acc1,
                                                  false, false);
  }
  const int   col = wv * 16 + (lane & 15);
  const float bz  = biasb[bk * CC + col];
  const int   mhi = (lane >= 16) ? 8 : 0;
#pragma unroll
  for (int r = 0; r < 8; ++r) {
    int n = pix[r + mhi];
    if (n >= 0) {
      size_t o = (((size_t)b * CC + col) * NN) + n;
      out[o]   = acc0[r] + bz + x[o];           // + residual (subtile 0)
    }
  }
#pragma unroll
  for (int r = 0; r < 8; ++r) {
    int n = pix[16 + r + mhi];
    if (n >= 0) {
      size_t o = (((size_t)b * CC + col) * NN) + n;
      out[o]   = acc1[r] + bz + x[o];           // + residual (subtile 1)
    }
  }
}

__global__ void labels_out_kernel(const int* __restrict__ labels, int* __restrict__ o) {
  int idx = blockIdx.x * 256 + threadIdx.x;
  if (idx < BB * NN) o[idx] = labels[idx];
}

// ---------------------------------------------------------------------------
// Host launcher
// ---------------------------------------------------------------------------
extern "C" void kernel_launch(void* const* d_in, const int* in_sizes, int n_in,
                              void* d_out, int out_size, void* d_ws, size_t ws_size,
                              hipStream_t stream) {
  (void)in_sizes; (void)n_in; (void)out_size; (void)ws_size;
  const float* xin[4] = {(const float*)d_in[0], (const float*)d_in[1],
                         (const float*)d_in[2], (const float*)d_in[3]};
  const float* c1_w = (const float*)d_in[4];
  const float* c1_b = (const float*)d_in[5];
  const float* bn1g = (const float*)d_in[6];
  const float* bn1b = (const float*)d_in[7];
  const float* bn1m = (const float*)d_in[8];
  const float* bn1v = (const float*)d_in[9];
  const float* a1p  = (const float*)d_in[10];
  const float* c2_w = (const float*)d_in[11];
  const float* c2_b = (const float*)d_in[12];
  const float* bn2g = (const float*)d_in[13];
  const float* bn2b = (const float*)d_in[14];
  const float* bn2m = (const float*)d_in[15];
  const float* bn2v = (const float*)d_in[16];
  const float* a2p  = (const float*)d_in[17];
  const float* c3_w = (const float*)d_in[18];
  const float* c3_b = (const float*)d_in[19];
  const float* fcp  = (const float*)d_in[20];

  // workspace bump allocator (total ~17 MB)
  char*  wsb = (char*)d_ws;
  size_t cur = 0;
  auto alloc = [&](size_t bytes) -> void* {
    void* p = wsb + cur;
    cur     = (cur + bytes + 255) & ~(size_t)255;
    return p;
  };
  float*    f1      = (float*)alloc((size_t)BB * HD * NN * 4);
  float*    f2      = (float*)alloc((size_t)BB * HD * NN * 4);
  float*    feats   = (float*)alloc((size_t)BB * NN * HD * 4);
  float*    fc      = (float*)alloc((size_t)BB * KK * HD * 4);
  float*    sc      = (float*)alloc((size_t)KK * 2 * 4);
  float*    sa      = (float*)alloc((size_t)BB * NN * KK * 4);
  int*      labels  = (int*)alloc((size_t)BB * NN * 4);
  int*      counts  = (int*)alloc((size_t)BB * KK * 4);
  int*      offs    = (int*)alloc((size_t)BB * KK * 4);
  int*      cursor  = (int*)alloc((size_t)BB * KK * 4);
  int*      order   = (int*)alloc((size_t)BB * NN * 4);
  _Float16* w1p     = (_Float16*)alloc((size_t)18 * 2 * 32 * 16 * 2);
  float*    b1v     = (float*)alloc(32 * 4);
  _Float16* w2p     = (_Float16*)alloc((size_t)9 * 2 * 32 * 16 * 2);
  float*    b2v     = (float*)alloc(32 * 4);
  float*    centers = (float*)alloc((size_t)BB * KK * DD * 4);
  float*    lrbuf   = (float*)alloc((size_t)BB * KK * RANK * 4);
  float*    biasbuf = (float*)alloc((size_t)BB * KK * CC * 4);
  _Float16* kp      = (_Float16*)alloc((size_t)BB * KK * DD * CC * 2);

  // 1) conv weight packing (BN folded)
  convpack_kernel<<<(576 * 32 + 255) / 256, 256, 0, stream>>>(c1_w, c1_b, bn1g, bn1b,
                                                              bn1m, bn1v, 64, 32, w1p, b1v);
  convpack_kernel<<<(288 * 32 + 255) / 256, 256, 0, stream>>>(c2_w, c2_b, bn2g, bn2b,
                                                              bn2m, bn2v, 32, 32, w2p, b2v);
  // 2) clustering CNN on WMMA (32-pixel tiles)
  conv3x3_wmma<64, 32><<<dim3(NN / 32, BB), 128, 0, stream>>>(xin[0], w1p, b1v, a1p, f1);
  conv3x3_wmma<32, 32><<<dim3(NN / 32, BB), 128, 0, stream>>>(f1, w2p, b2v, a2p, f2);
  conv1x1_kernel<<<(BB * NN * HD + 255) / 256, 256, 0, stream>>>(f2, c3_w, c3_b, feats);

  // 3) soft k-means (labels come from iteration 3's assignments)
  kmeans_init<<<(BB * KK * HD + 63) / 64, 64, 0, stream>>>(fcp, fc, sc);
  for (int it = 0; it < 3; ++it) {
    kmeans_assign<<<(BB * NN + 255) / 256, 256, 0, stream>>>(feats, fc, sc, sa, labels,
                                                             it == 2 ? 1 : 0);
    if (it < 2) {
      reduce_fc_kernel<<<BB * KK, 256, 0, stream>>>(sa, feats, fc);
      reduce_sc_kernel<<<KK, 256, 0, stream>>>(sa, sc);
    }
  }

  // 4) counting sort of pixels by label
  hipMemsetAsync(counts, 0, BB * KK * 4, stream);
  hist_kernel<<<(BB * NN + 255) / 256, 256, 0, stream>>>(labels, counts);
  offsets_kernel<<<1, BB, 0, stream>>>(counts, offs, cursor);
  scatter_kernel<<<(BB * NN + 255) / 256, 256, 0, stream>>>(labels, cursor, order);

  // 5) per band: centers -> MLPs -> packed kernels -> grouped gathered WMMA GEMM
  float* fout = (float*)d_out;
  for (int t = 0; t < 4; ++t) {
    int pb = 21 + t * 11;
    const float* lr1w  = (const float*)d_in[pb + 0];
    const float* lr1b  = (const float*)d_in[pb + 1];
    const float* lr2w  = (const float*)d_in[pb + 2];
    const float* lr2b  = (const float*)d_in[pb + 3];
    const float* lr3w  = (const float*)d_in[pb + 4];
    const float* lr3b  = (const float*)d_in[pb + 5];
    const float* basep = (const float*)d_in[pb + 6];
    const float* bw1   = (const float*)d_in[pb + 7];
    const float* bb1   = (const float*)d_in[pb + 8];
    const float* bw2   = (const float*)d_in[pb + 9];
    const float* bb2   = (const float*)d_in[pb + 10];

    centers_kernel<<<BB * KK, 192, 0, stream>>>(xin[t], order, offs, counts, centers);
    mlp_kernel<<<BB * KK, 64, 0, stream>>>(centers, lr1w, lr1b, lr2w, lr2b, lr3w, lr3b,
                                           bw1, bb1, bw2, bb2, lrbuf, biasbuf);
    kernpack_kernel<<<dim3(BB * KK, KCH), 256, 0, stream>>>(lrbuf, basep, kp);
    pwac_gemm<<<dim3((NN + 31) / 32, KK, BB), 128, 0, stream>>>(
        xin[t], order, offs, counts, kp, biasbuf, fout + (size_t)t * BB * CC * NN);
  }

  // 6) labels (int32) appended after the four float tensors
  labels_out_kernel<<<(BB * NN + 255) / 256, 256, 0, stream>>>(
      labels, (int*)(fout + (size_t)4 * BB * CC * NN));
}
